// ConvBnReLURes_23038204576398
// MI455X (gfx1250) — matrix operand order repaired and checked by reading, not compiled, not run on hardware
//
#include <hip/hip_runtime.h>
#include <hip/hip_bf16.h>
#include <stdint.h>

#define NPTS  60000
#define KNB   27
#define CCH   128
#define EPSV  1e-5f
#define SLOPE 0.01f

typedef __attribute__((ext_vector_type(16))) __bf16 v16bf;
typedef __attribute__((ext_vector_type(8)))  float  v8f;
typedef unsigned short u16;
typedef unsigned int   u32;

union FragBF { v16bf v; uint4 q[2]; };

__device__ __forceinline__ u16 f2bf(float f) {
    u32 u = __builtin_bit_cast(u32, f);
    u32 r = (u + 0x7FFFu + ((u >> 16) & 1u)) >> 16;
    return (u16)r;
}

// ---------------------------------------------------------------------------
// Sparse conv via WMMA + async gather-to-LDS.
// Block = 128 threads (4 waves), MTILE=64 points. Wave w owns 32 output
// channels (d0 = w*32). Triple-buffered LDS A-tiles filled with
// global_load_async_to_lds_b128 (per-lane gather), ASYNCcnt-pipelined with
// one barrier per k. nbr/mask preloaded to LDS so the async issue path is
// branchless (pointer select to a zero page for masked/OOB rows).
// Per-channel sum/sumsq accumulated into stats via global atomics.
// ---------------------------------------------------------------------------
#define MTILE 64
#define AROW  136   // LDS row stride in shorts (272B) to decorrelate banks

#define ASYNC_B128(off_str) \
    asm volatile("global_load_async_to_lds_b128 %0, %1, off offset:" off_str \
                 :: "v"(laddr), "v"(gaddr) : "memory")

__global__ __launch_bounds__(128)
void conv_wmma_bf16(const u16* __restrict__ xbf,
                    const int* __restrict__ nbr,
                    const unsigned char* __restrict__ mask,
                    const u16* __restrict__ Wbf,
                    const u16* __restrict__ zp,
                    float* __restrict__ out,
                    float* __restrict__ stats)
{
    __shared__ u16           At[3][MTILE * AROW];
    __shared__ int           sh_nbr[MTILE * KNB];
    __shared__ unsigned char sh_mask[MTILE * KNB];

    const int tid  = threadIdx.x;
    const int wave = tid >> 5;
    const int lane = tid & 31;
    const int row0 = blockIdx.x * MTILE;
    const int d0   = wave * 32;

    // ---- preload this block's nbr/mask slice (contiguous) into LDS ----
    for (int e = tid; e < MTILE * KNB; e += 128) {
        const int g = row0 * KNB + e;
        const bool valid = g < NPTS * KNB;
        sh_nbr[e]  = valid ? nbr[g]  : 0;
        sh_mask[e] = valid ? mask[g] : (unsigned char)0;
    }
    __syncthreads();

    // gather assignment: 128 threads, 64 rows, 2 threads per row (128B each)
    const int grow  = tid >> 1;
    const int ghalf = (tid & 1) * 128;      // byte offset within row

    // A-fragment addressing (16-bit A 16x32 layout)
    const int r    = lane & 15;
    const int klo  = (lane < 16) ? 0 : 8;
    const int half = lane >> 4;

    v8f acc[4][2];
    #pragma unroll
    for (int t = 0; t < 4; ++t) { acc[t][0] = (v8f){}; acc[t][1] = (v8f){}; }

    u16* pc = At[0];
    u16* pn = At[1];
    u16* pl = At[2];

    // branchless async gather issue for step k into lbuf
    auto issue_gather = [&](u16* lbuf, int k) {
        const int  idx = sh_nbr[grow * KNB + k];
        const bool m   = sh_mask[grow * KNB + k] != 0;
        const u16* src = m ? (xbf + (size_t)idx * CCH) : zp;
        const char* gaddr = (const char*)src + ghalf;
        const u32   laddr = (u32)(size_t)lbuf + (u32)(grow * (AROW * 2)) + (u32)ghalf;
        ASYNC_B128("0");  ASYNC_B128("16"); ASYNC_B128("32"); ASYNC_B128("48");
        ASYNC_B128("64"); ASYNC_B128("80"); ASYNC_B128("96"); ASYNC_B128("112");
    };

    issue_gather(pc, 0);

    for (int k = 0; k < KNB; ++k) {
        if (k + 1 < KNB) {
            issue_gather(pn, k + 1);
            asm volatile("s_wait_asynccnt 0x8" ::: "memory");  // batch k done
        } else {
            asm volatile("s_wait_asynccnt 0x0" ::: "memory");
        }
        __syncthreads();   // batch k visible to all waves

        const u16* Wk = Wbf + (size_t)k * (CCH * CCH);
        #pragma unroll
        for (int c0 = 0; c0 < CCH; c0 += 32) {
            // B fragments: lane = output col (lane&15); elements = K rows in
            // table order (klo+s, 16+klo+s-8), strided scalar loads from W
            FragBF B0, B1;
            const u16* wp = Wk + (size_t)c0 * CCH + d0 + (lane & 15);
            #pragma unroll
            for (int s = 0; s < 16; ++s) {
                const int kk = (s < 8) ? (klo + s) : (16 + klo + s - 8);
                B0.v[s] = __builtin_bit_cast(__bf16, wp[(size_t)kk * CCH]);
                B1.v[s] = __builtin_bit_cast(__bf16, wp[(size_t)kk * CCH + 16]);
            }

            #pragma unroll
            for (int t = 0; t < 4; ++t) {
                FragBF A;
                const u16* ap = pc + (t * 16 + r) * AROW + c0 + klo;
                A.q[0] = *(const uint4*)(ap);
                A.q[1] = *(const uint4*)(ap + 16);
                acc[t][0] = __builtin_amdgcn_wmma_f32_16x16x32_bf16(
                                false, A.v, false, B0.v, (short)0, acc[t][0], false, false);
                acc[t][1] = __builtin_amdgcn_wmma_f32_16x16x32_bf16(
                                false, A.v, false, B1.v, (short)0, acc[t][1], false, false);
            }
        }
        // rotate triple buffer
        u16* tp = pc; pc = pn; pn = pl; pl = tp;
    }

    // C/D layout: lane<16 -> col (lane), rows M = vgpr; lane>=16 -> M = 8+vgpr
    float s0 = 0.f, sq0 = 0.f, s1 = 0.f, sq1 = 0.f;
    #pragma unroll
    for (int t = 0; t < 4; ++t) {
        #pragma unroll
        for (int rr = 0; rr < 8; ++rr) {
            const int m = t * 16 + half * 8 + rr;
            const int n = row0 + m;
            const float v0 = acc[t][0][rr];
            const float v1 = acc[t][1][rr];
            if (n < NPTS) {
                out[(size_t)n * CCH + d0 + (lane & 15)]      = v0;
                out[(size_t)n * CCH + d0 + 16 + (lane & 15)] = v1;
            }
            s0 += v0; sq0 += v0 * v0;
            s1 += v1; sq1 += v1 * v1;
        }
    }
    atomicAdd(&stats[d0 + (lane & 15)],            s0);
    atomicAdd(&stats[CCH + d0 + (lane & 15)],      sq0);
    atomicAdd(&stats[d0 + 16 + (lane & 15)],       s1);
    atomicAdd(&stats[CCH + d0 + 16 + (lane & 15)], sq1);
}

// ---------------------------------------------------------------------------
// Streaming helpers (x4 vectorized; all element counts divide by 4)
// ---------------------------------------------------------------------------
__device__ __forceinline__ u32 pack2bf(float a, float b) {
    return (u32)f2bf(a) | ((u32)f2bf(b) << 16);
}

__global__ void cvt_f32_bf16_v4(const float4* __restrict__ src,
                                uint2* __restrict__ dst, int n4) {
    int i = blockIdx.x * blockDim.x + threadIdx.x;
    if (i < n4) {
        float4 f = src[i];
        dst[i] = make_uint2(pack2bf(f.x, f.y), pack2bf(f.z, f.w));
    }
}

__global__ void zero_f32(float* p, int n) {
    int i = blockIdx.x * blockDim.x + threadIdx.x;
    if (i < n) p[i] = 0.f;
}

__global__ void finalize_stats(const float* __restrict__ stats,
                               const float* __restrict__ gamma,
                               const float* __restrict__ beta,
                               float* __restrict__ sb)
{
    int c = threadIdx.x;   // 128 threads
    float mean  = stats[c] * (1.0f / NPTS);
    float var   = stats[CCH + c] * (1.0f / NPTS) - mean * mean;
    float inv   = rsqrtf(var + EPSV);
    float scale = gamma[c] * inv;
    sb[c]       = scale;
    sb[CCH + c] = beta[c] - mean * scale;
}

__device__ __forceinline__ float lrelu(float v) {
    return (v > 0.f) ? v : SLOPE * v;
}

__global__ void bn_lrelu_to_bf16_v4(const float4* __restrict__ h,
                                    const float* __restrict__ sb,
                                    uint2* __restrict__ outbf, int n4)
{
    int i = blockIdx.x * blockDim.x + threadIdx.x;
    if (i < n4) {
        const int c = (i * 4) & (CCH - 1);
        float4 v = h[i];
        v.x = lrelu(v.x * sb[c + 0] + sb[CCH + c + 0]);
        v.y = lrelu(v.y * sb[c + 1] + sb[CCH + c + 1]);
        v.z = lrelu(v.z * sb[c + 2] + sb[CCH + c + 2]);
        v.w = lrelu(v.w * sb[c + 3] + sb[CCH + c + 3]);
        outbf[i] = make_uint2(pack2bf(v.x, v.y), pack2bf(v.z, v.w));
    }
}

__global__ void bn_res_lrelu_v4(float4* __restrict__ h2,
                                const float4* __restrict__ x,
                                const float* __restrict__ sb, int n4)
{
    int i = blockIdx.x * blockDim.x + threadIdx.x;
    if (i < n4) {
        const int c = (i * 4) & (CCH - 1);
        float4 v = h2[i];
        float4 xr = x[i];
        v.x = lrelu(v.x * sb[c + 0] + sb[CCH + c + 0] + xr.x);
        v.y = lrelu(v.y * sb[c + 1] + sb[CCH + c + 1] + xr.y);
        v.z = lrelu(v.z * sb[c + 2] + sb[CCH + c + 2] + xr.z);
        v.w = lrelu(v.w * sb[c + 3] + sb[CCH + c + 3] + xr.w);
        h2[i] = v;
    }
}

// ---------------------------------------------------------------------------
extern "C" void kernel_launch(void* const* d_in, const int* in_sizes, int n_in,
                              void* d_out, int out_size, void* d_ws, size_t ws_size,
                              hipStream_t stream)
{
    (void)in_sizes; (void)n_in; (void)out_size; (void)ws_size;

    const float*         x      = (const float*)d_in[0];
    const int*           nbr    = (const int*)d_in[1];
    const unsigned char* mask   = (const unsigned char*)d_in[2];
    const float*         W1     = (const float*)d_in[3];
    const float*         W2     = (const float*)d_in[4];
    const float*         gamma1 = (const float*)d_in[5];
    const float*         beta1  = (const float*)d_in[6];
    const float*         gamma2 = (const float*)d_in[7];
    const float*         beta2  = (const float*)d_in[8];
    float*               out    = (float*)d_out;

    char* ws = (char*)d_ws;
    size_t off = 0;
    auto alloc = [&](size_t bytes) -> void* {
        void* p = ws + off;
        off = (off + bytes + 255) & ~(size_t)255;
        return p;
    };

    const int nx = NPTS * CCH;            // 7,680,000
    const int nw = KNB * CCH * CCH;       // 442,368

    u16*   xbf    = (u16*)  alloc((size_t)nx * 2);
    u16*   h1bf   = (u16*)  alloc((size_t)nx * 2);
    u16*   W1bf   = (u16*)  alloc((size_t)nw * 2);
    u16*   W2bf   = (u16*)  alloc((size_t)nw * 2);
    float* h1     = (float*)alloc((size_t)nx * 4);
    float* stats1 = (float*)alloc(2 * CCH * 4);
    float* stats2 = (float*)alloc(2 * CCH * 4);
    float* sb1    = (float*)alloc(2 * CCH * 4);
    float* sb2    = (float*)alloc(2 * CCH * 4);
    u16*   zp     = (u16*)  alloc(256);          // zero page for masked rows

    const int TB = 256;
    cvt_f32_bf16_v4<<<(nx / 4 + TB - 1) / TB, TB, 0, stream>>>(
        (const float4*)x, (uint2*)xbf, nx / 4);
    cvt_f32_bf16_v4<<<(nw / 4 + TB - 1) / TB, TB, 0, stream>>>(
        (const float4*)W1, (uint2*)W1bf, nw / 4);
    cvt_f32_bf16_v4<<<(nw / 4 + TB - 1) / TB, TB, 0, stream>>>(
        (const float4*)W2, (uint2*)W2bf, nw / 4);
    zero_f32<<<1, TB, 0, stream>>>(stats1, 2 * CCH);
    zero_f32<<<1, TB, 0, stream>>>(stats2, 2 * CCH);
    zero_f32<<<1, TB, 0, stream>>>((float*)zp, 64);

    const int cgrid = (NPTS + MTILE - 1) / MTILE;   // 938

    // conv1 -> h1 (fp32) + stats1
    conv_wmma_bf16<<<cgrid, 128, 0, stream>>>(xbf, nbr, mask, W1bf, zp, h1, stats1);
    finalize_stats<<<1, CCH, 0, stream>>>(stats1, gamma1, beta1, sb1);
    bn_lrelu_to_bf16_v4<<<(nx / 4 + TB - 1) / TB, TB, 0, stream>>>(
        (const float4*)h1, sb1, (uint2*)h1bf, nx / 4);

    // conv2 -> d_out (fp32 temp) + stats2
    conv_wmma_bf16<<<cgrid, 128, 0, stream>>>(h1bf, nbr, mask, W2bf, zp, out, stats2);
    finalize_stats<<<1, CCH, 0, stream>>>(stats2, gamma2, beta2, sb2);
    bn_res_lrelu_v4<<<(nx / 4 + TB - 1) / TB, TB, 0, stream>>>(
        (float4*)out, (const float4*)x, sb2, nx / 4);
}
